// PanelSegRetinaNet_71253507440703
// MI455X (gfx1250) — matrix operand order repaired and checked by reading, not compiled, not run on hardware
//
#include <hip/hip_runtime.h>
#include <hip/hip_bf16.h>

typedef __attribute__((ext_vector_type(16))) _Float16 v16h;
typedef __attribute__((ext_vector_type(8)))  _Float16 v8h;
typedef __attribute__((ext_vector_type(8)))  float    v8f;

#define NIMG   2
#define CCH    256
#define AANCH  3
#define CAP    1024
#define MAXDET 100

// Per-layer packed-weight size in halves: 16 co-tiles * 9 taps * 8 ci-blocks * (32 lanes * 16 halves)
#define FRAG_H   512
#define LAYER_H  ((size_t)16 * 9 * 8 * FRAG_H)

// LDS A-tile row stride in halves: 264 = 256 + 8 pad  (132 dwords, 132 % 64 == 4 -> 2-way max on b128 reads)
#define AROW   264
#define SPOS   32      // spatial positions per block (two 16-row WMMA sub-tiles)

// gfx1250 async global->LDS path (ASYNCcnt), with safe fallback to register staging.
#if defined(__has_builtin)
#if __has_builtin(__builtin_amdgcn_global_load_async_to_lds_b128) && \
    __has_builtin(__builtin_amdgcn_s_wait_asynccnt)
#define USE_ASYNC_LDS 1
#endif
#endif
#ifndef USE_ASYNC_LDS
#define USE_ASYNC_LDS 0
#endif

#if USE_ASYNC_LDS
// Builtin signature (from hipcc diagnostic): both pointers are to 128-bit int vectors,
// global side in addrspace(1) ("__device__"), LDS side in addrspace(3).
typedef int a128_t __attribute__((vector_size(4 * sizeof(int))));
#define ASYNC_CP_B128(gsrc, ldst)                                          \
    __builtin_amdgcn_global_load_async_to_lds_b128(                        \
        (__attribute__((address_space(1))) a128_t*)(gsrc),                 \
        (__attribute__((address_space(3))) a128_t*)(ldst), 0, 0)
#endif

// ---------------- weight pack: OIHW f32 -> WMMA B-fragment-ready f16 ----------------
// Fragment layout (per 16-co tile, per tap, per 32-ci block):
//   lane L<16 : N = L,     K(j) = j<8 ? j : 8 + j       (j = 0..15)
//   lane L>=16: N = L-16,  K(j) = j<8 ? 8 + j : 16 + j
__global__ void pack_weights_kernel(const float* __restrict__ cls_w,
                                    const float* __restrict__ box_w,
                                    _Float16* __restrict__ wpack, int total)
{
    int t = blockIdx.x * blockDim.x + threadIdx.x;
    if (t >= total) return;
    int u = t;
    int j    = u & 15; u >>= 4;
    int lane = u & 31; u >>= 5;
    int cb   = u & 7;  u >>= 3;
    int kyx  = u % 9;  u /= 9;
    int cot  = u & 15; u >>= 4;
    int l    = u;                       // 0..3 cls tower, 4..7 box tower
    int n  = lane & 15;
    int kk = (j < 8) ? (((lane >> 4) << 3) + j) : (8 + ((lane >> 4) << 3) + j);
    int co = cot * 16 + n;
    int ci = cb * 32 + kk;
    const float* src = (l < 4) ? cls_w : box_w;
    int ll = (l < 4) ? l : (l - 4);
    float w = src[(((size_t)ll * CCH + co) * CCH + ci) * 9 + kyx];
    wpack[t] = (_Float16)w;
}

// ---------------- feature convert: NCHW f32 -> padded NHWC f16 ----------------
__global__ void convert_pad_kernel(const float* __restrict__ feat,
                                   _Float16* __restrict__ dst,
                                   int H, int W, int total)
{
    int t = blockIdx.x * blockDim.x + threadIdx.x;
    if (t >= total) return;
    int Wp = W + 2, Hp = H + 2;
    int u = t;
    int ci = u & 255; u >>= 8;
    int xp = u % Wp;  u /= Wp;
    int yp = u % Hp;
    int img = u / Hp;
    _Float16 v = (_Float16)0.f;
    if (xp > 0 && yp > 0 && xp <= W && yp <= H)
        v = (_Float16)feat[(((size_t)img * CCH + ci) * H + (yp - 1)) * W + (xp - 1)];
    dst[t] = v;
}

// ---------------- conv3x3 256->256 + bias + ReLU via WMMA implicit GEMM ----------------
// Block = 256 threads = 8 waves, computes 32 spatial positions x 256 output channels.
// wave wv: spatial sub-tile st = wv>>2 (16 rows), channel slab cot0 = (wv&3)*4 (64 channels,
// 4 accumulators). Per tap: the 32x256 f16 A tile is staged in LDS. With the gfx1250 async
// path, tap t+1 streams global->LDS via GLOBAL_LOAD_ASYNC_TO_LDS (double-buffered, ASYNCcnt)
// while tap t computes; otherwise chunks are prefetched through registers. Each wave feeds
// one A fragment (2x ds_read_b128) into 4 WMMAs per ci-block.
__global__ __launch_bounds__(256)
void conv3x3_wmma_kernel(const _Float16* __restrict__ in,     // padded NHWC f16 (NIMG images)
                         const _Float16* __restrict__ wpack,  // this layer's packed weights
                         const float* __restrict__ bias,      // 256
                         _Float16* __restrict__ out,          // padded NHWC f16
                         int H, int W, int nTiles)
{
#if USE_ASYNC_LDS
    __shared__ _Float16 sA[2][SPOS * AROW];   // 33792 B, double-buffered taps
#else
    __shared__ _Float16 sA1[SPOS * AROW];     // 16896 B
#endif

    const int HW = H * W;
    const int Wp = W + 2;
    const int tile = blockIdx.x % nTiles;
    const int img  = blockIdx.x / nTiles;
    const int tid  = threadIdx.x;
    const int lane = tid & 31;
    const int wv   = tid >> 5;
    const int st   = wv >> 2;              // spatial sub-tile 0/1
    const int cot0 = (wv & 3) * 4;         // four consecutive 16-co tiles
    const int hi   = lane >> 4;            // half-wave selects K sub-ranges
    const int m    = lane & 15;            // A-matrix row within sub-tile
    const int aoff0 = hi ? 8 : 0;          // ISA 16-bit A 16x32 K layout
    const int aoff1 = hi ? 24 : 16;

    const _Float16* inp = in + (size_t)img * (H + 2) * Wp * CCH;

    // ---- cooperative-load chunk mapping: 32 rows x 32 chunks of 8 halves = 1024 chunks, 4/thread
    int gbase[4], lbase[4];
#pragma unroll
    for (int k = 0; k < 4; ++k) {
        const int ch = tid + k * 256;
        const int rr = ch >> 5;            // 0..31
        const int c8 = (ch & 31) * 8;      // ci offset in halves
        int sr = tile * SPOS + rr; if (sr >= HW) sr = HW - 1;
        const int yr = sr / W, xr = sr - yr * W;
        gbase[k] = (yr * Wp + xr) * CCH + c8;   // tap (ky,kx) adds (ky*Wp+kx)*CCH
        lbase[k] = rr * AROW + c8;
    }

    v8f acc[4];
#pragma unroll
    for (int cc = 0; cc < 4; ++cc) acc[cc] = (v8f){};

#if USE_ASYNC_LDS
    // prologue: async-copy tap 0 into buffer 0
#pragma unroll
    for (int k = 0; k < 4; ++k)
        ASYNC_CP_B128(inp + gbase[k], &sA[0][lbase[k]]);

    for (int kyx = 0; kyx < 9; ++kyx) {
        __builtin_amdgcn_s_wait_asynccnt(0);   // this tap's tile landed in LDS
        __syncthreads();                       // ...for every wave; prior tap's reads done

        if (kyx < 8) {                         // stream next tap into the other buffer
            const int ky = (kyx + 1) / 3, kx = (kyx + 1) - ky * 3;
            const int d = (ky * Wp + kx) * CCH;
            _Float16* dstbuf = sA[(kyx + 1) & 1];
#pragma unroll
            for (int k = 0; k < 4; ++k)
                ASYNC_CP_B128(inp + gbase[k] + d, dstbuf + lbase[k]);
        }

        const _Float16* arow = sA[kyx & 1] + (size_t)(st * 16 + m) * AROW;
#pragma unroll
        for (int cb = 0; cb < 8; ++cb) {
            union { v16h v; v8h h[2]; } a;
            a.h[0] = *(const v8h*)(arow + cb * 32 + aoff0);
            a.h[1] = *(const v8h*)(arow + cb * 32 + aoff1);
#pragma unroll
            for (int cc = 0; cc < 4; ++cc) {
                v16h b = *(const v16h*)(wpack + ((size_t)((cot0 + cc) * 9 + kyx) * 8 + cb) * FRAG_H
                                        + lane * 16);
                acc[cc] = __builtin_amdgcn_wmma_f32_16x16x32_f16(false, a.v, false, b,
                                                                 (short)0, acc[cc], false, false);
            }
        }
    }
#else
    // fallback: stage chunks through registers (prefetch next tap before compute)
    v8h g[4];
#pragma unroll
    for (int k = 0; k < 4; ++k)
        g[k] = *(const v8h*)(inp + gbase[k]);

    for (int kyx = 0; kyx < 9; ++kyx) {
        __syncthreads();                   // previous tap's ds reads complete
#pragma unroll
        for (int k = 0; k < 4; ++k)
            *(v8h*)(sA1 + lbase[k]) = g[k];
        __syncthreads();                   // stores visible

        if (kyx < 8) {                     // prefetch next tap while computing this one
            const int ky = (kyx + 1) / 3, kx = (kyx + 1) - ky * 3;
            const int d = (ky * Wp + kx) * CCH;
#pragma unroll
            for (int k = 0; k < 4; ++k)
                g[k] = *(const v8h*)(inp + gbase[k] + d);
        }

        const _Float16* arow = sA1 + (size_t)(st * 16 + m) * AROW;
#pragma unroll
        for (int cb = 0; cb < 8; ++cb) {
            union { v16h v; v8h h[2]; } a;
            a.h[0] = *(const v8h*)(arow + cb * 32 + aoff0);
            a.h[1] = *(const v8h*)(arow + cb * 32 + aoff1);
#pragma unroll
            for (int cc = 0; cc < 4; ++cc) {
                v16h b = *(const v16h*)(wpack + ((size_t)((cot0 + cc) * 9 + kyx) * 8 + cb) * FRAG_H
                                        + lane * 16);
                acc[cc] = __builtin_amdgcn_wmma_f32_16x16x32_f16(false, a.v, false, b,
                                                                 (short)0, acc[cc], false, false);
            }
        }
    }
#endif

    float bs[4];
#pragma unroll
    for (int cc = 0; cc < 4; ++cc) bs[cc] = bias[(cot0 + cc) * 16 + m];

#pragma unroll
    for (int v = 0; v < 8; ++v) {
        const int M  = hi * 8 + v;         // C/D layout: lanes<16 rows 0..7, lanes>=16 rows 8..15
        const int so = tile * SPOS + st * 16 + M;
        if (so < HW) {
            const int yo = so / W, xo = so - yo * W;
            const size_t base = ((size_t)img * (H + 2) * Wp + (size_t)(yo + 1) * Wp + (xo + 1)) * CCH;
#pragma unroll
            for (int cc = 0; cc < 4; ++cc) {
                float r = acc[cc][v] + bs[cc];
                r = r > 0.f ? r : 0.f;
                out[base + (cot0 + cc) * 16 + m] = (_Float16)r;
            }
        }
    }
}

// ---------------- small head conv (OC = 3 or 12), fp32 out, no activation ----------------
__global__ void head_conv_kernel(const _Float16* __restrict__ in, const float* __restrict__ w,
                                 const float* __restrict__ b, float* __restrict__ out,
                                 int H, int W, int OC, int total)
{
    int t = blockIdx.x * blockDim.x + threadIdx.x;
    if (t >= total) return;
    const int HW = H * W, Wp = W + 2;
    const int oc  = t % OC;
    const int s   = (t / OC) % HW;
    const int img = t / (OC * HW);
    const int y = s / W, x = s - y * W;
    float acc = b[oc];
    for (int kyx = 0; kyx < 9; ++kyx) {
        const int ky = kyx / 3, kx = kyx - ky * 3;
        const _Float16* row = in + ((size_t)img * (H + 2) * Wp + (size_t)(y + ky) * Wp + (x + kx)) * CCH;
        const float* wr = w + (size_t)oc * CCH * 9 + kyx;
        for (int ci = 0; ci < CCH; ++ci)
            acc = fmaf((float)row[ci], wr[(size_t)ci * 9], acc);
    }
    out[((size_t)img * HW + s) * OC + oc] = acc;
}

// ---------------- sigmoid + threshold + anchor decode -> candidate list ----------------
__global__ void candidates_kernel(const float* __restrict__ logits, const float* __restrict__ regs,
                                  float* __restrict__ cand, int* __restrict__ cnt,
                                  int H, int W, float stride, float sz, int total)
{
    int t = blockIdx.x * blockDim.x + threadIdx.x;
    if (t >= total) return;
    const int HW = H * W;
    const int a   = t % AANCH;
    const int s   = (t / AANCH) % HW;
    const int img = t / (AANCH * HW);
    const float lg = logits[((size_t)img * HW + s) * AANCH + a];
    const float sc = 1.f / (1.f + expf(-lg));
    if (sc <= 0.05f) return;
    const int y = s / W, x = s - y * W;
    const float r  = (a == 0) ? 0.5f : (a == 1 ? 1.f : 2.f);
    const float wa = sqrtf(sz * sz / r);
    const float ha = wa * r;
    const float cxa = (x + 0.5f) * stride;
    const float cya = (y + 0.5f) * stride;
    const float* d = regs + ((size_t)img * HW + s) * 12 + a * 4;
    const float SCALE_CLAMP = 4.135166556742356f;   // log(1000/16)
    const float dx = d[0], dy = d[1];
    const float dw = fminf(d[2], SCALE_CLAMP), dh = fminf(d[3], SCALE_CLAMP);
    const float cx = dx * wa + cxa, cy = dy * ha + cya;
    const float bw = expf(dw) * wa, bh = expf(dh) * ha;
    const float x1 = fminf(fmaxf(cx - 0.5f * bw, 0.f), 800.f);
    const float y1 = fminf(fmaxf(cy - 0.5f * bh, 0.f), 800.f);
    const float x2 = fminf(fmaxf(cx + 0.5f * bw, 0.f), 800.f);
    const float y2 = fminf(fmaxf(cy + 0.5f * bh, 0.f), 800.f);
    const int pos = atomicAdd(&cnt[img], 1);
    if (pos < CAP) {
        float* c = cand + ((size_t)img * CAP + pos) * 5;
        c[0] = x1; c[1] = y1; c[2] = x2; c[3] = y2; c[4] = sc;
    }
}

// ---------------- LDS-staged greedy NMS, one block per image ----------------
__global__ __launch_bounds__(256)
void nms_kernel(const float* __restrict__ cand, const int* __restrict__ cnt,
                float* __restrict__ out)
{
    __shared__ float sx1[CAP], sy1[CAP], sx2[CAP], sy2[CAP], ssc[CAP];
    __shared__ float rbest[256];
    __shared__ int   ridx[256];
    __shared__ float bsel[5];
    __shared__ int   jsel;
    const int img = blockIdx.x;
    const int tid = threadIdx.x;
    int n = cnt[img]; if (n > CAP) n = CAP;
    for (int i = tid; i < n; i += 256) {
        const float* c = cand + ((size_t)img * CAP + i) * 5;
        sx1[i] = c[0]; sy1[i] = c[1]; sx2[i] = c[2]; sy2[i] = c[3]; ssc[i] = c[4];
    }
    __syncthreads();
    for (int it = 0; it < MAXDET; ++it) {
        float best = 0.f; int bj = -1;
        for (int i = tid; i < n; i += 256)
            if (ssc[i] > best) { best = ssc[i]; bj = i; }   // ascending -> first max kept
        rbest[tid] = best; ridx[tid] = bj;
        __syncthreads();
        if (tid == 0) {
            float bb = 0.f; int jj = -1;
            for (int k = 0; k < 256; ++k) {
                if (ridx[k] < 0) continue;
                if (rbest[k] > bb || (rbest[k] == bb && jj >= 0 && ridx[k] < jj)) {
                    bb = rbest[k]; jj = ridx[k];
                }
            }
            float* o = out + ((size_t)img * MAXDET + it) * 5;
            if (jj >= 0 && bb > 0.f) {
                jsel = jj;
                bsel[0] = sx1[jj]; bsel[1] = sy1[jj];
                bsel[2] = sx2[jj]; bsel[3] = sy2[jj]; bsel[4] = bb;
                o[0] = bsel[0]; o[1] = bsel[1]; o[2] = bsel[2]; o[3] = bsel[3]; o[4] = bb;
                ssc[jj] = 0.f;
            } else {
                jsel = -1;
                o[0] = o[1] = o[2] = o[3] = o[4] = 0.f;
            }
        }
        __syncthreads();
        if (jsel >= 0) {
            const float bx1 = bsel[0], by1 = bsel[1], bx2 = bsel[2], by2 = bsel[3];
            const float a1 = (bx2 - bx1) * (by2 - by1);
            for (int i = tid; i < n; i += 256) {
                const float ltx = fmaxf(bx1, sx1[i]), lty = fmaxf(by1, sy1[i]);
                const float rbx = fminf(bx2, sx2[i]), rby = fminf(by2, sy2[i]);
                const float iw = fmaxf(rbx - ltx, 0.f), ih = fmaxf(rby - lty, 0.f);
                const float inter = iw * ih;
                const float a2 = (sx2[i] - sx1[i]) * (sy2[i] - sy1[i]);
                const float iou = inter / (a1 + a2 - inter + 1e-9f);
                if (iou > 0.5f) ssc[i] = 0.f;
            }
        }
        __syncthreads();
    }
}

// =======================================================================================
extern "C" void kernel_launch(void* const* d_in, const int* in_sizes, int n_in,
                              void* d_out, int out_size, void* d_ws, size_t ws_size,
                              hipStream_t stream)
{
    const float* feat[5];
    for (int i = 0; i < 5; ++i) feat[i] = (const float*)d_in[i];
    const float* cls_w   = (const float*)d_in[5];
    const float* cls_b   = (const float*)d_in[6];
    const float* box_w   = (const float*)d_in[7];
    const float* box_b   = (const float*)d_in[8];
    const float* score_w = (const float*)d_in[9];
    const float* score_b = (const float*)d_in[10];
    const float* pred_w  = (const float*)d_in[11];
    const float* pred_b  = (const float*)d_in[12];

    static const int   Hs[5]      = {100, 50, 25, 13, 7};
    static const int   Ws_[5]     = {100, 50, 25, 13, 7};
    static const float strides[5] = {8.f, 16.f, 32.f, 64.f, 128.f};
    static const float sizes[5]   = {32.f, 64.f, 128.f, 256.f, 512.f};

    // ---- carve workspace ----
    char* ws = (char*)d_ws;
    size_t cur = 0;
    auto alloc = [&](size_t bytes) -> char* {
        char* p = ws + cur;
        cur += (bytes + 255) & ~(size_t)255;
        return p;
    };
    _Float16* wpack = (_Float16*)alloc((size_t)8 * LAYER_H * sizeof(_Float16));
    size_t padStart = cur;
    _Float16 *bufF[5], *bufA[5], *bufB[5];
    for (int l = 0; l < 5; ++l) {
        size_t e = (size_t)NIMG * (Hs[l] + 2) * (Ws_[l] + 2) * CCH * sizeof(_Float16);
        bufF[l] = (_Float16*)alloc(e);
        bufA[l] = (_Float16*)alloc(e);
        bufB[l] = (_Float16*)alloc(e);
    }
    size_t padEnd = cur;
    float *logits[5], *regs[5];
    for (int l = 0; l < 5; ++l) {
        logits[l] = (float*)alloc((size_t)NIMG * Hs[l] * Ws_[l] * AANCH * sizeof(float));
        regs[l]   = (float*)alloc((size_t)NIMG * Hs[l] * Ws_[l] * 12 * sizeof(float));
    }
    float* cand = (float*)alloc((size_t)NIMG * CAP * 5 * sizeof(float));
    int*   cnt  = (int*)alloc(NIMG * sizeof(int));

    // ---- zero padded buffers (borders), counters, output ----
    (void)hipMemsetAsync(ws + padStart, 0, padEnd - padStart, stream);
    (void)hipMemsetAsync(cnt, 0, NIMG * sizeof(int), stream);
    (void)hipMemsetAsync(d_out, 0, (size_t)out_size * sizeof(float), stream);

    // ---- pack tower weights to fragment layout ----
    {
        int total = (int)(8 * LAYER_H);
        pack_weights_kernel<<<(total + 255) / 256, 256, 0, stream>>>(cls_w, box_w, wpack, total);
    }

    // ---- convert features ----
    for (int l = 0; l < 5; ++l) {
        int total = NIMG * (Hs[l] + 2) * (Ws_[l] + 2) * CCH;
        convert_pad_kernel<<<(total + 255) / 256, 256, 0, stream>>>(feat[l], bufF[l], Hs[l], Ws_[l], total);
    }

    // ---- towers + heads + candidates per level ----
    for (int l = 0; l < 5; ++l) {
        const int H = Hs[l], W = Ws_[l], HW = H * W;
        const int nTiles = (HW + SPOS - 1) / SPOS;
        const dim3 cgrid(NIMG * nTiles);

        // cls tower: layers 0..3
        conv3x3_wmma_kernel<<<cgrid, 256, 0, stream>>>(bufF[l], wpack + 0 * LAYER_H, cls_b + 0 * CCH, bufA[l], H, W, nTiles);
        conv3x3_wmma_kernel<<<cgrid, 256, 0, stream>>>(bufA[l], wpack + 1 * LAYER_H, cls_b + 1 * CCH, bufB[l], H, W, nTiles);
        conv3x3_wmma_kernel<<<cgrid, 256, 0, stream>>>(bufB[l], wpack + 2 * LAYER_H, cls_b + 2 * CCH, bufA[l], H, W, nTiles);
        conv3x3_wmma_kernel<<<cgrid, 256, 0, stream>>>(bufA[l], wpack + 3 * LAYER_H, cls_b + 3 * CCH, bufB[l], H, W, nTiles);
        {
            int total = NIMG * HW * AANCH;
            head_conv_kernel<<<(total + 255) / 256, 256, 0, stream>>>(bufB[l], score_w, score_b, logits[l], H, W, AANCH, total);
        }

        // box tower: layers 4..7
        conv3x3_wmma_kernel<<<cgrid, 256, 0, stream>>>(bufF[l], wpack + 4 * LAYER_H, box_b + 0 * CCH, bufA[l], H, W, nTiles);
        conv3x3_wmma_kernel<<<cgrid, 256, 0, stream>>>(bufA[l], wpack + 5 * LAYER_H, box_b + 1 * CCH, bufB[l], H, W, nTiles);
        conv3x3_wmma_kernel<<<cgrid, 256, 0, stream>>>(bufB[l], wpack + 6 * LAYER_H, box_b + 2 * CCH, bufA[l], H, W, nTiles);
        conv3x3_wmma_kernel<<<cgrid, 256, 0, stream>>>(bufA[l], wpack + 7 * LAYER_H, box_b + 3 * CCH, bufB[l], H, W, nTiles);
        {
            int total = NIMG * HW * 12;
            head_conv_kernel<<<(total + 255) / 256, 256, 0, stream>>>(bufB[l], pred_w, pred_b, regs[l], H, W, 12, total);
        }

        {
            int total = NIMG * HW * AANCH;
            candidates_kernel<<<(total + 255) / 256, 256, 0, stream>>>(logits[l], regs[l], cand, cnt,
                                                                       H, W, strides[l], sizes[l], total);
        }
    }

    // ---- NMS -> (2, 100, 5) ----
    nms_kernel<<<NIMG, 256, 0, stream>>>(cand, cnt, (float*)d_out);
}
